// WindowAttention_41815801594647
// MI455X (gfx1250) — compile-verified, matrix operand order checked by reading, not run on hardware
//
#include <hip/hip_runtime.h>

// ---------------------------------------------------------------------------
// Window attention, fused per-window kernel for gfx1250 (CDNA5, wave32, WMMA)
// B=4096, N=49 (padded to 64), C=128, H=4, DH=32
// K-reductions split into dual independent WMMA chains to avoid the
// WMMA->WMMA hazard NOPs (ISA 7.12.1).
// ---------------------------------------------------------------------------

typedef __attribute__((ext_vector_type(16))) _Float16 v16h;
typedef __attribute__((ext_vector_type(8)))  float    v8f;

namespace cfg {
constexpr int NTOK = 49;      // real tokens per window
constexpr int NPAD = 64;      // padded tokens (4 tiles of 16)
constexpr int CH   = 128;     // channels
constexpr int NH   = 4;       // heads
constexpr int DH   = 32;      // head dim
constexpr int C3   = 3 * CH;  // qkv output width = 384
constexpr float SCALE = 0.17677669529663687f;  // 1/sqrt(32)

// LDS strides (in halfs), padded to dodge bank conflicts
constexpr int XSTR = CH + 8;    // 136, x / attn-out buffer
constexpr int QSTR = C3 + 8;    // 392, qkv buffer
constexpr int PSTR = NPAD + 8;  // 72,  per-wave softmax(P) slab
constexpr int XS_OFF  = 0;
constexpr int QKV_OFF = NPAD * XSTR;                 // 8704
constexpr int PS_OFF  = QKV_OFF + NPAD * QSTR;       // 33792
constexpr int LDS_HALFS = PS_OFF + 8 * 16 * PSTR;    // 43008 halfs = 84 KB
}

// --- WMMA fragment helpers (16x16x32 f16, wave32 VGPR layouts per ISA 7.12.2)
// kk pattern per VGPR i: i<4 -> 2i ; i>=4 -> 16+2(i-4); lanes>=16 add +8.

__device__ __forceinline__ int frag_kk(int i) {
    return (i < 4) ? 2 * i : 16 + 2 * (i - 4);
}

// A fragment, or B fragment when K runs along the contiguous dim of src
// (i.e. src is row-major [M][K] for A, or [N][K] == B^T for B).
__device__ __forceinline__ v16h ldfrag_kcontig(const _Float16* src, int row0,
                                               int stride, int colbase) {
    const int lane = threadIdx.x & 31;
    const int mn   = lane & 15;
    const int khi  = (lane >> 4) ? 8 : 0;
    const _Float16* p = src + (row0 + mn) * stride + colbase + khi;
    v16h f;
#pragma unroll
    for (int i = 0; i < 8; ++i) {
        const int kk = frag_kk(i);
        f[2 * i]     = p[kk];
        f[2 * i + 1] = p[kk + 1];
    }
    return f;
}

// B fragment from a row-major [K][N] source (strided k-pair loads).
__device__ __forceinline__ v16h ldfrag_b_rowmajor(const _Float16* src, int k0,
                                                  int n0, int stride) {
    const int lane = threadIdx.x & 31;
    const int n    = lane & 15;
    const int khi  = (lane >> 4) ? 8 : 0;
    const _Float16* p = src + n0 + n;
    v16h f;
#pragma unroll
    for (int i = 0; i < 8; ++i) {
        const int kk = k0 + khi + frag_kk(i);
        f[2 * i]     = p[(size_t)kk * stride];
        f[2 * i + 1] = p[(size_t)(kk + 1) * stride];
    }
    return f;
}

// B fragment from pair-packed weights: element (k,n) at ((k>>1)*nout+n)*2+(k&1)
// -> each VGPR is one contiguous 32-bit (half2) load.
__device__ __forceinline__ v16h ldfrag_b_packed(const _Float16* w, int k0,
                                                int n0, int nout) {
    const int lane = threadIdx.x & 31;
    const int n    = lane & 15;
    const int khi  = (lane >> 4) ? 8 : 0;
    v16h f;
#pragma unroll
    for (int i = 0; i < 8; ++i) {
        const int kk = k0 + khi + frag_kk(i);
        const _Float16* p = w + ((size_t)(kk >> 1) * nout + n0 + n) * 2;
        f[2 * i]     = p[0];
        f[2 * i + 1] = p[1];
    }
    return f;
}

// Store a 16x16 f32 accumulator tile to LDS as f16 (C/D layout: VGPR r = row,
// lanes 16-31 rows +8, N = lane%16).
__device__ __forceinline__ void st_d_f16(_Float16* dst, int row0, int col0,
                                         int stride, v8f acc) {
    const int lane  = threadIdx.x & 31;
    const int n     = lane & 15;
    const int rbase = (lane >> 4) ? 8 : 0;
#pragma unroll
    for (int r = 0; r < 8; ++r)
        dst[(row0 + rbase + r) * stride + col0 + n] = (_Float16)acc[r];
}

__device__ __forceinline__ v8f wmma_f16(v16h a, v16h b, v8f c) {
    return __builtin_amdgcn_wmma_f32_16x16x32_f16(
        /*neg_a=*/false, a, /*neg_b=*/false, b,
        /*c_mod=*/(short)0, c, /*reuse_a=*/false, /*reuse_b=*/false);
}

__device__ __forceinline__ v8f v8f_add(v8f a, v8f b) {
    v8f r;
#pragma unroll
    for (int i = 0; i < 8; ++i) r[i] = a[i] + b[i];
    return r;
}

// ---------------------------------------------------------------------------
// Prep: convert fp32 weights to pair-packed f16 once (weights then sit in L2).
// ---------------------------------------------------------------------------
__global__ void WindowAttention_prep_weights(const float* __restrict__ wq,
                                             const float* __restrict__ wp,
                                             _Float16* __restrict__ wq16,
                                             _Float16* __restrict__ wp16) {
    const int tid = blockIdx.x * blockDim.x + threadIdx.x;
    if (tid < cfg::CH * cfg::C3) {
        const int k = tid / cfg::C3, n = tid % cfg::C3;
        wq16[((k >> 1) * cfg::C3 + n) * 2 + (k & 1)] = (_Float16)wq[tid];
    } else if (tid < cfg::CH * cfg::C3 + cfg::CH * cfg::CH) {
        const int t = tid - cfg::CH * cfg::C3;
        const int k = t / cfg::CH, n = t % cfg::CH;
        wp16[((k >> 1) * cfg::CH + n) * 2 + (k & 1)] = (_Float16)wp[t];
    }
}

// ---------------------------------------------------------------------------
// Main fused kernel: one workgroup (8 waves) per window.
// ---------------------------------------------------------------------------
__global__ void __launch_bounds__(256)
WindowAttention_41815801594647_kernel(const float* __restrict__ x,
                                      const _Float16* __restrict__ wq16,
                                      const _Float16* __restrict__ wp16,
                                      const float* __restrict__ bp,
                                      float* __restrict__ out) {
    using namespace cfg;
    extern __shared__ _Float16 smem[];
    _Float16* xs  = smem + XS_OFF;   // [NPAD][XSTR] : x (f16), later attn-out
    _Float16* qkv = smem + QKV_OFF;  // [NPAD][QSTR] : q|k|v (f16)
    _Float16* ps  = smem + PS_OFF;   // [8][16][PSTR]: per-wave softmax(P) slab

    const int tid  = threadIdx.x;
    const int wave = tid >> 5;
    const int lane = tid & 31;
    const int blk  = blockIdx.x;

    // hint L2 for this block's weight stream
    __builtin_prefetch(wq16, 0, 1);

    // ---- Phase 1: stage x into LDS as f16, zero the padded rows -----------
    {
        const float4* xv = (const float4*)(x + (size_t)blk * NTOK * CH);
        for (int idx = tid; idx < NTOK * (CH / 4); idx += 256) {
            const int row = idx / (CH / 4);
            const int c4  = idx % (CH / 4);
            const float4 v = xv[idx];
            _Float16* d = xs + row * XSTR + c4 * 4;
            d[0] = (_Float16)v.x; d[1] = (_Float16)v.y;
            d[2] = (_Float16)v.z; d[3] = (_Float16)v.w;
        }
        for (int idx = tid; idx < (NPAD - NTOK) * CH; idx += 256) {
            const int row = NTOK + idx / CH;
            xs[row * XSTR + (idx % CH)] = (_Float16)0.0f;
        }
    }
    __syncthreads();

    // ---- Phase 2: qkv = x @ Wqkv (64x384 = 4 x 24 tiles, K=128) -----------
    // Dual accumulator chains (k even / k odd) -> independent WMMA streams.
    for (int t = wave; t < 4 * (C3 / 16); t += 8) {
        const int mt = t / (C3 / 16);
        const int nt = t % (C3 / 16);
        v8f acc0 = {}, acc1 = {};
#pragma unroll
        for (int kp = 0; kp < 2; ++kp) {
            v16h a0 = ldfrag_kcontig(xs, mt * 16, XSTR, (2 * kp + 0) * 32);
            v16h b0 = ldfrag_b_packed(wq16, (2 * kp + 0) * 32, nt * 16, C3);
            v16h a1 = ldfrag_kcontig(xs, mt * 16, XSTR, (2 * kp + 1) * 32);
            v16h b1 = ldfrag_b_packed(wq16, (2 * kp + 1) * 32, nt * 16, C3);
            acc0 = wmma_f16(a0, b0, acc0);
            acc1 = wmma_f16(a1, b1, acc1);
        }
        st_d_f16(qkv, mt * 16, nt * 16, QSTR, v8f_add(acc0, acc1));
    }
    __syncthreads();

    // ---- Phase 3: per-head attention. Task = (head, 16-row block). --------
    _Float16* ao = xs;  // x staging no longer needed -> reuse as attn output
    _Float16* pw = ps + wave * 16 * PSTR;
#pragma unroll
    for (int it = 0; it < 2; ++it) {
        const int task = wave + 8 * it;
        const int h    = task >> 2;
        const int mt   = task & 3;

        // scores S = q @ k^T (16 x 64, K = DH = 32): 4 independent WMMAs
        v8f st[4];
        {
            v16h aq = ldfrag_kcontig(qkv, mt * 16, QSTR, h * DH);
#pragma unroll
            for (int nt = 0; nt < 4; ++nt) {
                v8f z = {};
                v16h bk = ldfrag_kcontig(qkv, nt * 16, QSTR, CH + h * DH);
                st[nt] = wmma_f16(aq, bk, z);
            }
        }

        // scale + mask padded key columns
        const int cin = lane & 15;
#pragma unroll
        for (int nt = 0; nt < 4; ++nt) {
            const bool ok = (nt * 16 + cin) < NTOK;
#pragma unroll
            for (int r = 0; r < 8; ++r)
                st[nt][r] = ok ? st[nt][r] * SCALE : -3.0e38f;
        }

        // row-wise softmax: reduce across the 16-lane group + 4 tiles
#pragma unroll
        for (int r = 0; r < 8; ++r) {
            float m = fmaxf(fmaxf(st[0][r], st[1][r]), fmaxf(st[2][r], st[3][r]));
#pragma unroll
            for (int off = 8; off >= 1; off >>= 1)
                m = fmaxf(m, __shfl_xor(m, off, 16));
            float s = 0.0f;
#pragma unroll
            for (int nt = 0; nt < 4; ++nt) {
                const float e = __expf(st[nt][r] - m);
                st[nt][r] = e;
                s += e;
            }
#pragma unroll
            for (int off = 8; off >= 1; off >>= 1)
                s += __shfl_xor(s, off, 16);
            const float inv = 1.0f / s;
#pragma unroll
            for (int nt = 0; nt < 4; ++nt) st[nt][r] *= inv;
        }

        // P -> f16 into this wave's private LDS slab (wave-local RAW;
        // backend orders via s_wait_dscnt — no workgroup barrier needed)
#pragma unroll
        for (int nt = 0; nt < 4; ++nt)
            st_d_f16(pw, 0, nt * 16, PSTR, st[nt]);

        // out = P(16x64) @ V(64x32): per n-tile, 2 fully independent chains
#pragma unroll
        for (int dt = 0; dt < 2; ++dt) {
            v8f z0 = {}, z1 = {};
            v16h a0 = ldfrag_kcontig(pw, 0, PSTR, 0);
            v16h b0 = ldfrag_b_rowmajor(qkv, 0, 2 * CH + h * DH + dt * 16, QSTR);
            v16h a1 = ldfrag_kcontig(pw, 0, PSTR, 32);
            v16h b1 = ldfrag_b_rowmajor(qkv, 32, 2 * CH + h * DH + dt * 16, QSTR);
            v8f acc0 = wmma_f16(a0, b0, z0);
            v8f acc1 = wmma_f16(a1, b1, z1);
            st_d_f16(ao, mt * 16, h * DH + dt * 16, XSTR, v8f_add(acc0, acc1));
        }
    }
    __syncthreads();

    // ---- Phase 4: proj + bias, fp32 store of the 49 real rows -------------
    for (int t = wave; t < 4 * (CH / 16); t += 8) {
        const int mt = t >> 3;
        const int nt = t & 7;
        v8f acc0 = {}, acc1 = {};
#pragma unroll
        for (int kp = 0; kp < 2; ++kp) {
            v16h a0 = ldfrag_kcontig(ao, mt * 16, XSTR, (2 * kp + 0) * 32);
            v16h b0 = ldfrag_b_packed(wp16, (2 * kp + 0) * 32, nt * 16, CH);
            v16h a1 = ldfrag_kcontig(ao, mt * 16, XSTR, (2 * kp + 1) * 32);
            v16h b1 = ldfrag_b_packed(wp16, (2 * kp + 1) * 32, nt * 16, CH);
            acc0 = wmma_f16(a0, b0, acc0);
            acc1 = wmma_f16(a1, b1, acc1);
        }
        const v8f acc = v8f_add(acc0, acc1);
        const int col  = nt * 16 + (lane & 15);
        const float bias = bp[col];
        const int rbase = (lane >> 4) ? 8 : 0;
#pragma unroll
        for (int r = 0; r < 8; ++r) {
            const int m = mt * 16 + rbase + r;
            if (m < NTOK)
                out[((size_t)blk * NTOK + m) * CH + col] = acc[r] + bias;
        }
    }
}

// ---------------------------------------------------------------------------
extern "C" void kernel_launch(void* const* d_in, const int* in_sizes, int n_in,
                              void* d_out, int out_size, void* d_ws,
                              size_t ws_size, hipStream_t stream) {
    using namespace cfg;
    const float* x      = (const float*)d_in[0];  // (4096, 49, 128)
    const float* w_qkv  = (const float*)d_in[1];  // (128, 384)
    const float* w_proj = (const float*)d_in[2];  // (128, 128)
    const float* b_proj = (const float*)d_in[3];  // (128,)
    float* out = (float*)d_out;                   // (4096, 49, 128)

    _Float16* wq16 = (_Float16*)d_ws;             // 49152 halfs, pair-packed
    _Float16* wp16 = wq16 + CH * C3;              // 16384 halfs, pair-packed

    const int prep_elems = CH * C3 + CH * CH;     // 65536
    WindowAttention_prep_weights<<<(prep_elems + 255) / 256, 256, 0, stream>>>(
        w_qkv, w_proj, wq16, wp16);

    const size_t lds_bytes = (size_t)LDS_HALFS * sizeof(_Float16);  // 84 KB
    hipFuncSetAttribute((const void*)WindowAttention_41815801594647_kernel,
                        hipFuncAttributeMaxDynamicSharedMemorySize,
                        (int)lds_bytes);

    const int B = in_sizes[0] / (NTOK * CH);      // 4096
    WindowAttention_41815801594647_kernel<<<B, 256, lds_bytes, stream>>>(
        x, wq16, wp16, b_proj, out);
}
